// DOAModel_88776974008365
// MI455X (gfx1250) — matrix operand order repaired and checked by reading, not compiled, not run on hardware
//
#include <hip/hip_runtime.h>
#include <hip/hip_bf16.h>
#include <math.h>

// Problem constants (from reference): b=16, c=896, h=w=40 -> N=1600, TOPK=20
#define BATCH 16
#define CDIM  896
#define NPIX  1600
#define TOPK  20

typedef __attribute__((ext_vector_type(16))) __bf16 v16bf;
typedef __attribute__((ext_vector_type(8)))  float  v8f;

union Frag { uint4 u[2]; v16bf v; };

// ---------------------------------------------------------------------------
// 0) zero small stat buffers (rowsum/colsum) -- re-run every call (graph replay)
// ---------------------------------------------------------------------------
__global__ void zero_kernel(float* __restrict__ p, int n) {
    int i = blockIdx.x * blockDim.x + threadIdx.x;
    if (i < n) p[i] = 0.0f;
}

// ---------------------------------------------------------------------------
// 1) inverse L2 norm per (b, pixel) over the 896 channels.
//    x layout: [b][c][p], p fastest -> fully coalesced across threads.
// ---------------------------------------------------------------------------
__global__ __launch_bounds__(256)
void invnorm_kernel(const float* __restrict__ x, float* __restrict__ invn) {
    int t = blockIdx.x * 256 + threadIdx.x;          // 0 .. 25599
    int b = t / NPIX, p = t % NPIX;
    const float* xb = x + (size_t)b * CDIM * NPIX + p;
    float s = 0.0f;
    #pragma unroll 4
    for (int c = 0; c < CDIM; ++c) {
        float v = xb[(size_t)c * NPIX];
        s += v * v;
    }
    invn[t] = 1.0f / fmaxf(sqrtf(s), 1e-12f);        // matches F.normalize eps
}

// ---------------------------------------------------------------------------
// 2) normalize + transpose [c][p] -> [p][c] + convert to bf16 (LDS 32x33 tile)
//    Output is row-major [N, c] so WMMA fragments load as plain B128 vectors.
// ---------------------------------------------------------------------------
__global__ __launch_bounds__(256)
void norm_transpose(const float* __restrict__ x, const float* __restrict__ invn,
                    __hip_bfloat16* __restrict__ xt) {
    __shared__ float t[32][33];
    const int b  = blockIdx.z;
    const int p0 = blockIdx.x * 32;                  // 50 tiles (1600)
    const int c0 = blockIdx.y * 32;                  // 28 tiles (896)
    const float* xb = x + (size_t)b * CDIM * NPIX;
    #pragma unroll
    for (int l = 0; l < 4; ++l) {
        int lin = threadIdx.x + l * 256;
        int ci = lin >> 5, pi = lin & 31;
        t[ci][pi] = xb[(size_t)(c0 + ci) * NPIX + p0 + pi] * invn[b * NPIX + p0 + pi];
    }
    __syncthreads();
    __hip_bfloat16* xtb = xt + (size_t)b * NPIX * CDIM;
    #pragma unroll
    for (int l = 0; l < 4; ++l) {
        int lin = threadIdx.x + l * 256;
        int pi = lin >> 5, ci = lin & 31;
        xtb[(size_t)(p0 + pi) * CDIM + c0 + ci] = __float2bfloat16(t[ci][pi]);
    }
}

// ---------------------------------------------------------------------------
// 3) Batched NT-GEMM via v_wmma_f32_16x16x32_bf16 + fused softmax-stats
//    epilogue.  Block = 128 threads = 4 waves, block tile 64x64, each wave
//    owns a 32x32 tile (2x2 WMMA 16x16 tiles), K loop = 896/32 = 28 iters.
//    Stores e = exp(alpha * dot) into the xc_o_q output region and
//    atomically accumulates row sums (sum over q) and col sums (sum over p).
//    |logit| <= ~10  =>  exp is fp32-safe without max subtraction.
//
//    A frag (16x32 bf16, ISA 7.12.2): lane half h=lane>>4 holds K runs
//      [h*8, h*8+8) and [h*8+16, h*8+24)  -> two 16B loads per row.
//    B frag (32x16 bf16, same K order as A): lane half h holds K runs
//      [h*8, h*8+8) and [h*8+16, h*8+24)  -> two 16B loads per column.
// ---------------------------------------------------------------------------
__global__ __launch_bounds__(128)
void gemm_exp_stats(const __hip_bfloat16* __restrict__ At,
                    const __hip_bfloat16* __restrict__ Bt,
                    const float* __restrict__ alphap,
                    float* __restrict__ outE,
                    float* __restrict__ rowsum,
                    float* __restrict__ colsum) {
    const int b    = blockIdx.z;
    const int tid  = threadIdx.x;
    const int wave = tid >> 5, lane = tid & 31;
    const int half = lane >> 4, l15 = lane & 15;
    const int wr = wave >> 1, wc = wave & 1;
    const int m0 = blockIdx.x * 64 + wr * 32;
    const int n0 = blockIdx.y * 64 + wc * 32;

    const __hip_bfloat16* A  = At + (size_t)b * NPIX * CDIM;
    const __hip_bfloat16* Bm = Bt + (size_t)b * NPIX * CDIM;
    const __hip_bfloat16* ra0 = A  + (size_t)(m0 +      l15) * CDIM;
    const __hip_bfloat16* ra1 = A  + (size_t)(m0 + 16 + l15) * CDIM;
    const __hip_bfloat16* rb0 = Bm + (size_t)(n0 +      l15) * CDIM;
    const __hip_bfloat16* rb1 = Bm + (size_t)(n0 + 16 + l15) * CDIM;

    v8f acc00 = {}, acc01 = {}, acc10 = {}, acc11 = {};
    const int aoff = half * 8;    // elements
    const int boff = half * 8;    // elements

    for (int k = 0; k < CDIM; k += 32) {
        Frag fa0, fa1, fb0, fb1;
        fa0.u[0] = *(const uint4*)(ra0 + k + aoff);
        fa0.u[1] = *(const uint4*)(ra0 + k + aoff + 16);
        fa1.u[0] = *(const uint4*)(ra1 + k + aoff);
        fa1.u[1] = *(const uint4*)(ra1 + k + aoff + 16);
        fb0.u[0] = *(const uint4*)(rb0 + k + boff);
        fb0.u[1] = *(const uint4*)(rb0 + k + boff + 16);
        fb1.u[0] = *(const uint4*)(rb1 + k + boff);
        fb1.u[1] = *(const uint4*)(rb1 + k + boff + 16);
        acc00 = __builtin_amdgcn_wmma_f32_16x16x32_bf16(false, fa0.v, false, fb0.v, (short)0, acc00, false, false);
        acc01 = __builtin_amdgcn_wmma_f32_16x16x32_bf16(false, fa0.v, false, fb1.v, (short)0, acc01, false, false);
        acc10 = __builtin_amdgcn_wmma_f32_16x16x32_bf16(false, fa1.v, false, fb0.v, (short)0, acc10, false, false);
        acc11 = __builtin_amdgcn_wmma_f32_16x16x32_bf16(false, fa1.v, false, fb1.v, (short)0, acc11, false, false);
    }

    const float alpha = *alphap;
    #pragma unroll
    for (int r = 0; r < 8; ++r) {
        acc00[r] = __expf(acc00[r] * alpha);
        acc01[r] = __expf(acc01[r] * alpha);
        acc10[r] = __expf(acc10[r] * alpha);
        acc11[r] = __expf(acc11[r] * alpha);
    }

    // store e (C/D layout: VGPR r -> row r (+8 for upper half-lanes), col = l15)
    float* outb = outE + (size_t)b * NPIX * NPIX;
    #pragma unroll
    for (int r = 0; r < 8; ++r) {
        int mrow0 = m0 +      r + half * 8;
        int mrow1 = m0 + 16 + r + half * 8;
        outb[(size_t)mrow0 * NPIX + n0 +      l15] = acc00[r];
        outb[(size_t)mrow0 * NPIX + n0 + 16 + l15] = acc01[r];
        outb[(size_t)mrow1 * NPIX + n0 +      l15] = acc10[r];
        outb[(size_t)mrow1 * NPIX + n0 + 16 + l15] = acc11[r];
    }

    // row sums: reduce across the 16 lanes of each half, one atomic per row
    #pragma unroll
    for (int r = 0; r < 8; ++r) {
        float s0 = acc00[r] + acc01[r];
        float s1 = acc10[r] + acc11[r];
        #pragma unroll
        for (int m = 1; m < 16; m <<= 1) {
            s0 += __shfl_xor(s0, m, 32);
            s1 += __shfl_xor(s1, m, 32);
        }
        if (l15 == 0) {
            atomicAdd(&rowsum[b * NPIX + m0 +      r + half * 8], s0);
            atomicAdd(&rowsum[b * NPIX + m0 + 16 + r + half * 8], s1);
        }
    }

    // col sums: fold the two halves (same columns), one atomic per column
    float c0 = 0.0f, c1 = 0.0f;
    #pragma unroll
    for (int r = 0; r < 8; ++r) {
        c0 += acc00[r] + acc10[r];
        c1 += acc01[r] + acc11[r];
    }
    c0 += __shfl_xor(c0, 16, 32);
    c1 += __shfl_xor(c1, 16, 32);
    if (half == 0) {
        atomicAdd(&colsum[b * NPIX + n0 +      l15], c0);
        atomicAdd(&colsum[b * NPIX + n0 + 16 + l15], c1);
    }
}

// ---------------------------------------------------------------------------
// 4) finalize: x_c = e^2 / (rowsum * colsum); rewrite xc_o_q in place and
//    write the transpose into xc_o_p through an LDS 32x33 tile.
// ---------------------------------------------------------------------------
__global__ __launch_bounds__(256)
void finalize_transpose(float* __restrict__ xcq, float* __restrict__ xcp,
                        const float* __restrict__ rowsum,
                        const float* __restrict__ colsum) {
    __shared__ float t[32][33];
    __shared__ float rs[32], cs[32];
    const int b  = blockIdx.z;
    const int p0 = blockIdx.x * 32;
    const int q0 = blockIdx.y * 32;
    if (threadIdx.x < 32)       rs[threadIdx.x]      = rowsum[b * NPIX + p0 + threadIdx.x];
    else if (threadIdx.x < 64)  cs[threadIdx.x - 32] = colsum[b * NPIX + q0 + threadIdx.x - 32];
    __syncthreads();
    float* baseq = xcq + (size_t)b * NPIX * NPIX;
    float* basep = xcp + (size_t)b * NPIX * NPIX;
    #pragma unroll
    for (int l = 0; l < 4; ++l) {
        int lin = threadIdx.x + l * 256;
        int pi = lin >> 5, qi = lin & 31;
        size_t idx = (size_t)(p0 + pi) * NPIX + q0 + qi;
        float e = baseq[idx];
        float v = (e * e) / (rs[pi] * cs[qi]);
        baseq[idx] = v;
        t[pi][qi] = v;
    }
    __syncthreads();
    #pragma unroll
    for (int l = 0; l < 4; ++l) {
        int lin = threadIdx.x + l * 256;
        int qi = lin >> 5, pi = lin & 31;
        basep[(size_t)(q0 + qi) * NPIX + p0 + pi] = t[pi][qi];
    }
}

// ---------------------------------------------------------------------------
// 5) top-20 (descending) per contiguous row of 1600: iterative max extraction.
//    out[b][k][p] layout.  One block of 128 threads per row.
// ---------------------------------------------------------------------------
__global__ __launch_bounds__(128)
void topk_rows(const float* __restrict__ mat, float* __restrict__ out) {
    __shared__ float buf[NPIX];
    __shared__ float rv[128];
    __shared__ int   ri[128];
    const int row = blockIdx.x;                 // b*NPIX + p
    const int b = row / NPIX, p = row % NPIX;
    const float* src = mat + (size_t)row * NPIX;
    for (int i = threadIdx.x; i < NPIX; i += 128) buf[i] = src[i];
    __syncthreads();
    for (int it = 0; it < TOPK; ++it) {
        float best = -INFINITY; int bi = threadIdx.x;
        for (int i = threadIdx.x; i < NPIX; i += 128) {
            float v = buf[i];
            if (v > best) { best = v; bi = i; }
        }
        rv[threadIdx.x] = best; ri[threadIdx.x] = bi;
        __syncthreads();
        for (int s = 64; s > 0; s >>= 1) {
            if (threadIdx.x < s && rv[threadIdx.x + s] > rv[threadIdx.x]) {
                rv[threadIdx.x] = rv[threadIdx.x + s];
                ri[threadIdx.x] = ri[threadIdx.x + s];
            }
            __syncthreads();
        }
        if (threadIdx.x == 0) {
            out[((size_t)b * TOPK + it) * NPIX + p] = rv[0];
            buf[ri[0]] = -INFINITY;
        }
        __syncthreads();
    }
}

// ---------------------------------------------------------------------------
extern "C" void kernel_launch(void* const* d_in, const int* in_sizes, int n_in,
                              void* d_out, int out_size, void* d_ws, size_t ws_size,
                              hipStream_t stream) {
    const float* xp    = (const float*)d_in[0];   // [16,896,40,40]
    const float* xq    = (const float*)d_in[1];   // [16,896,40,40]
    const float* alpha = (const float*)d_in[2];   // scalar

    float* out  = (float*)d_out;
    float* valp = out;                                    // [16,20,40,40]
    float* valq = out + (size_t)BATCH * TOPK * NPIX;      // [16,20,40,40]
    float* xcp  = valq + (size_t)BATCH * TOPK * NPIX;     // [16,1600,40,40]
    float* xcq  = xcp  + (size_t)BATCH * NPIX * NPIX;     // [16,1600,40,40]

    // workspace layout (~92.2 MB)
    char* ws = (char*)d_ws;
    __hip_bfloat16* xpt = (__hip_bfloat16*)ws;                       // [16][1600][896] bf16
    __hip_bfloat16* xqt = (__hip_bfloat16*)(ws + (size_t)BATCH * NPIX * CDIM * 2);
    float* rowsum = (float*)(ws + (size_t)2 * BATCH * NPIX * CDIM * 2);
    float* colsum = rowsum + BATCH * NPIX;
    float* invp   = colsum + BATCH * NPIX;
    float* invq   = invp   + BATCH * NPIX;

    // 0) zero stats (rowsum + colsum contiguous)
    zero_kernel<<<(2 * BATCH * NPIX + 255) / 256, 256, 0, stream>>>(rowsum, 2 * BATCH * NPIX);

    // 1) per-pixel inverse L2 norms
    invnorm_kernel<<<(BATCH * NPIX) / 256, 256, 0, stream>>>(xp, invp);
    invnorm_kernel<<<(BATCH * NPIX) / 256, 256, 0, stream>>>(xq, invq);

    // 2) normalize + transpose + bf16 pack
    dim3 gT(NPIX / 32, CDIM / 32, BATCH);
    norm_transpose<<<gT, 256, 0, stream>>>(xp, invp, xpt);
    norm_transpose<<<gT, 256, 0, stream>>>(xq, invq, xqt);

    // 3) WMMA GEMM + exp + softmax denominators (e stored into xcq region)
    dim3 gG(NPIX / 64, NPIX / 64, BATCH);
    gemm_exp_stats<<<gG, 128, 0, stream>>>(xpt, xqt, alpha, xcq, rowsum, colsum);

    // 4) finalize dual-softmax product + transposed copy
    dim3 gF(NPIX / 32, NPIX / 32, BATCH);
    finalize_transpose<<<gF, 256, 0, stream>>>(xcq, xcp, rowsum, colsum);

    // 5) top-20: valp from rows of x_c (xcq), valq from rows of x_c^T (xcp)
    topk_rows<<<BATCH * NPIX, 128, 0, stream>>>(xcq, valp);
    topk_rows<<<BATCH * NPIX, 128, 0, stream>>>(xcp, valq);
}
